// CrossSpectralSpatialAttention_14413910245925
// MI455X (gfx1250) — matrix operand order repaired and checked by reading, not compiled, not run on hardware
//
#include <hip/hip_runtime.h>
#include <cmath>

// ---------------------------------------------------------------------------
// CrossSpectralSpatialAttention for MI455X (gfx1250, wave32).
// v_wmma_f32_16x16x32_f16 for all matmuls, f16 intermediates,
// global_load_async_to_lds_b128 staging, vectorized LDS fragment loads.
//
// B=1, C=128, D=31, H=W=32 -> P = 31744 positions, p = d*1024 + (h*32+w).
// x layout: x[c*P + p] (f32). Intermediates: row-major f16 [p][c].
// ---------------------------------------------------------------------------

#define PPOS 31744
#define CDIM 128

typedef __attribute__((ext_vector_type(16))) _Float16 v16h;
typedef __attribute__((ext_vector_type(8)))  _Float16 v8h;
typedef __attribute__((ext_vector_type(4)))  _Float16 v4h;
typedef __attribute__((ext_vector_type(8)))  float    v8f;

// gfx1250 WMMA 16x16x32 f16 layouts (CDNA5 ISA §7.12.2):
//  A: element e of lane -> m = lane&15, k = 16*(e>=8)+2*((e>>1)&3)+8*(lane>>4)+(e&1)
//     => per lane: two contiguous runs k in [8h,8h+8) and [16+8h,16+8h+8).
//  B: element e -> n = lane&15, k as for A; this code uses k = 16*(lane>>4)+e in A and B alike.
//  C/D: element r -> m = r + 8*(lane>>4), n = lane&15.
__device__ __forceinline__ int b_k(int lane, int e) {
  return ((lane >> 4) << 4) + e;
}

__device__ __forceinline__ v16h zero16h() {
  v16h z; for (int i = 0; i < 16; ++i) z[i] = (_Float16)0.0f; return z;
}
__device__ __forceinline__ v8h zero8h() {
  v8h z; for (int i = 0; i < 8; ++i) z[i] = (_Float16)0.0f; return z;
}
__device__ __forceinline__ v4h zero4h() {
  v4h z; for (int i = 0; i < 4; ++i) z[i] = (_Float16)0.0f; return z;
}
__device__ __forceinline__ v16h cat8(v8h lo, v8h hi) {
  return __builtin_shufflevector(lo, hi, 0, 1, 2, 3, 4, 5, 6, 7,
                                 8, 9, 10, 11, 12, 13, 14, 15);
}

// Async global->LDS copy of 16 bytes (gfx1250 Tensor/async path, ASYNCcnt).
__device__ __forceinline__ void async_cp16(unsigned lds_off,
                                           const _Float16* g) {
  asm volatile("global_load_async_to_lds_b128 %0, %1, off"
               :: "v"(lds_off), "v"(g) : "memory");
}
__device__ __forceinline__ void async_wait() {
  asm volatile("s_wait_asynccnt 0" ::: "memory");
}
__device__ __forceinline__ unsigned lds_off(const void* p) {
  return (unsigned)(uintptr_t)p;  // low 32 bits of flat LDS address = offset
}

// ---------------------------------------------------------------------------
// x (f32, [c][p]) -> xh (f16, [p][c]) transpose+convert via LDS tile.
// ---------------------------------------------------------------------------
__global__ __launch_bounds__(256) void x_to_f16_kernel(
    const float* __restrict__ x, _Float16* __restrict__ xh) {
  __shared__ _Float16 T[64][36];
  const int p0 = blockIdx.x * 64, c0 = blockIdx.y * 32;
  const int t = threadIdx.x;
  for (int i = 0; i < 8; ++i) {
    int e = t + 256 * i;
    int pl = e & 63, cl = e >> 6;
    T[pl][cl] = (_Float16)x[(size_t)(c0 + cl) * PPOS + p0 + pl];
  }
  __syncthreads();
  for (int i = 0; i < 2; ++i) {
    int e = t + 256 * i;
    int row = e >> 3, seg = e & 7;
    *(v4h*)&xh[(size_t)(p0 + row) * CDIM + c0 + seg * 4] =
        *(const v4h*)&T[row][seg * 4];
  }
}

// ---------------------------------------------------------------------------
// GEMM: Out[m][n] = sum_k A16[m*128+k] * W[k*128+n] + bias[n]
// A async-staged to LDS; W staged transposed (Wt[n][k]) so B fragments are
// single contiguous 32B LDS loads. 128 rows/block, 8 waves x 16 rows.
// ---------------------------------------------------------------------------
template <bool F16OUT>
__global__ __launch_bounds__(256) void gemm128_kernel(
    const _Float16* __restrict__ A, const float* __restrict__ W,
    const float* __restrict__ bias, void* __restrict__ OutP) {
  __shared__ _Float16 As[128][136];  // rows 272B (16B-multiple)
  __shared__ _Float16 Wt[128][144];  // rows 288B (32B-multiple)
  const int t = threadIdx.x;
  const int m0 = blockIdx.x * 128;
  const unsigned asBase = lds_off(&As[0][0]);

  for (int i = 0; i < 8; ++i) {  // 2048 16B chunks: A tile 128x128 f16
    int c = t + 256 * i;
    int m = c >> 4, seg = c & 15;
    async_cp16(asBase + m * 272 + seg * 16,
               A + (size_t)(m0 + m) * CDIM + seg * 8);
  }
  for (int i = 0; i < 64; ++i) {  // W f32 -> f16 transposed
    int idx = t + 256 * i;
    int k = idx >> 7, n = idx & 127;
    Wt[n][k] = (_Float16)W[idx];
  }
  async_wait();
  __syncthreads();

  const int w = t >> 5, lane = t & 31, half = lane >> 4, l16 = lane & 15;
  const int mrow = w * 16 + l16;
  v16h af[4];
  for (int kt = 0; kt < 4; ++kt) {
    v8h lo = *(const v8h*)&As[mrow][kt * 32 + 16 * half];
    v8h hi = *(const v8h*)&As[mrow][kt * 32 + 16 * half + 8];
    af[kt] = cat8(lo, hi);
  }
  for (int nt = 0; nt < 8; ++nt) {
    float bv = bias[nt * 16 + l16];
    v8f acc;
    for (int r = 0; r < 8; ++r) acc[r] = bv;
    for (int kt = 0; kt < 4; ++kt) {
      v16h bf = *(const v16h*)&Wt[nt * 16 + l16][kt * 32 + 16 * half];
      acc = __builtin_amdgcn_wmma_f32_16x16x32_f16(false, af[kt], false, bf,
                                                   (short)0, acc, false, false);
    }
    for (int r = 0; r < 8; ++r) {
      int m = w * 16 + r + 8 * half;
      size_t o = (size_t)(m0 + m) * CDIM + nt * 16 + l16;
      if (F16OUT) ((_Float16*)OutP)[o] = (_Float16)acc[r];
      else        ((float*)OutP)[o]    = acc[r];
    }
  }
}

// ---------------------------------------------------------------------------
// Spectral attention: 1024 seqs (spatial site s), L=31 (pad 32), 8 heads,
// d=16. Block = sequence, wave = head. logits = q.q/16 (SCALE^2 in A frag).
// ---------------------------------------------------------------------------
__global__ __launch_bounds__(256) void spectral_attn_kernel(
    const _Float16* __restrict__ Q, _Float16* __restrict__ O) {
  __shared__ _Float16 Qs[32][144];     // rows 288B; row 31 zeroed
  __shared__ float    Sw[8][32][33];
  __shared__ _Float16 Pw[8][32][48];   // rows 96B (16B-multiple)
  const int s = blockIdx.x;
  const int t = threadIdx.x;
  const unsigned qsBase = lds_off(&Qs[0][0]);

  for (int i = 0; i < 2; ++i) {  // 31 rows x 16 chunks = 496 async copies
    int c = t + 256 * i;
    if (c < 496) {
      int l = c >> 4, seg = c & 15;
      async_cp16(qsBase + l * 288 + seg * 16,
                 Q + (size_t)(l * 1024 + s) * CDIM + seg * 8);
    }
  }
  if (t < 36) {  // zero pad row 31
    v4h z4 = zero4h();
    *(v4h*)&Qs[31][t * 4] = z4;
  }
  async_wait();
  __syncthreads();

  const int h = t >> 5, lane = t & 31, half = lane >> 4, l16 = lane & 15;
  v8f z;
  for (int r = 0; r < 8; ++r) z[r] = 0.0f;

  // S = (Q/16) Q^T, K = 16 padded to 32 (upper half of A and B zero).
  v16h aS[2], bS[2];
  for (int mt = 0; mt < 2; ++mt) {
    v8h lo = *(const v8h*)&Qs[mt * 16 + l16][h * 16];
    v8h hi = *(const v8h*)&Qs[mt * 16 + l16][h * 16 + 8];
    for (int e = 0; e < 8; ++e) { lo[e] = lo[e] * (_Float16)0.0625f; hi[e] = hi[e] * (_Float16)0.0625f; }
    aS[mt] = half ? zero16h() : cat8(lo, hi);
  }
  for (int nt = 0; nt < 2; ++nt) {
    v16h row = *(const v16h*)&Qs[nt * 16 + l16][h * 16];
    bS[nt] = half ? zero16h() : row;  // half1 lanes carry k>=16 -> zero
  }
  for (int mt = 0; mt < 2; ++mt)
    for (int nt = 0; nt < 2; ++nt) {
      v8f sv = __builtin_amdgcn_wmma_f32_16x16x32_f16(false, aS[mt], false,
                                                      bS[nt], (short)0, z,
                                                      false, false);
      for (int r = 0; r < 8; ++r)
        Sw[h][mt * 16 + r + 8 * half][nt * 16 + l16] = sv[r];
    }
  __syncthreads();

  {  // row softmax over keys 0..30 (key 31 masked)
    int row = lane;
    float mx = -3.0e38f;
    for (int j = 0; j < 31; ++j) mx = fmaxf(mx, Sw[h][row][j]);
    float sum = 0.0f;
    for (int j = 0; j < 31; ++j) {
      float p = __expf(Sw[h][row][j] - mx);
      Sw[h][row][j] = p;
      sum += p;
    }
    float inv = 1.0f / sum;
    for (int j = 0; j < 31; ++j) Pw[h][row][j] = (_Float16)(Sw[h][row][j] * inv);
    Pw[h][row][31] = (_Float16)0.0f;
  }
  __syncthreads();

  // O = P V (K = 32 keys); V columns gathered from Qs (small, scalar).
  v16h bV;
  for (int e = 0; e < 16; ++e) bV[e] = Qs[b_k(lane, e)][h * 16 + l16];
  for (int mt = 0; mt < 2; ++mt) {
    v8h lo = *(const v8h*)&Pw[h][mt * 16 + l16][16 * half];
    v8h hi = *(const v8h*)&Pw[h][mt * 16 + l16][16 * half + 8];
    v16h aP = cat8(lo, hi);
    v8f ov = __builtin_amdgcn_wmma_f32_16x16x32_f16(false, aP, false, bV,
                                                    (short)0, z, false, false);
    for (int r = 0; r < 8; ++r) {
      int m = mt * 16 + r + 8 * half;
      if (m < 31)
        O[((size_t)(m * 1024 + s)) * CDIM + h * 16 + l16] = (_Float16)ov[r];
    }
  }
}

// ---------------------------------------------------------------------------
// Spatial flash attention: 31 seqs x 8 heads, L=1024, d=16.
// Grid (qtile=8, head=8, seq=31); wave owns 16 q-rows; online softmax with
// xor-shuffle row reductions (masks 1..8 stay inside each 16-lane half).
// ---------------------------------------------------------------------------
__global__ __launch_bounds__(256) void spatial_flash_kernel(
    const _Float16* __restrict__ Q, _Float16* __restrict__ O) {
  __shared__ _Float16 Kt[32][16];    // [key][dim], rows 32B
  __shared__ _Float16 Ktt[16][32];   // [dim][key], rows 64B
  __shared__ _Float16 Pw[8][16][48]; // per-wave P tile, rows 96B
  const int qt = blockIdx.x, h = blockIdx.y, n = blockIdx.z;
  const int t = threadIdx.x, w = t >> 5, lane = t & 31;
  const int half = lane >> 4, l16 = lane & 15;
  const size_t seqBase = (size_t)n * 1024;
  const unsigned ktBase = lds_off(&Kt[0][0]);

  // Q fragment (16 rows), scaled by SCALE^2 = 1/16; k>=16 zero.
  const int qrow = qt * 128 + w * 16 + l16;
  v16h aQ;
  {
    v8h lo = *(const v8h*)&Q[(seqBase + qrow) * CDIM + h * 16];
    v8h hi = *(const v8h*)&Q[(seqBase + qrow) * CDIM + h * 16 + 8];
    for (int e = 0; e < 8; ++e) { lo[e] = lo[e] * (_Float16)0.0625f; hi[e] = hi[e] * (_Float16)0.0625f; }
    aQ = half ? zero16h() : cat8(lo, hi);
  }

  float rmax[8], rsum[8];
  v8f oacc;
  for (int r = 0; r < 8; ++r) { rmax[r] = -3.0e38f; rsum[r] = 0.0f; oacc[r] = 0.0f; }

  for (int kt = 0; kt < 32; ++kt) {
    __syncthreads();
    if (t < 64) {  // async stage 32 keys x 16 dims (64 x 16B)
      int key = t >> 1, sh = t & 1;
      async_cp16(ktBase + key * 32 + sh * 16,
                 Q + (seqBase + kt * 32 + key) * CDIM + h * 16 + sh * 8);
    }
    async_wait();
    __syncthreads();
    for (int i = 0; i < 2; ++i) {  // build transposed copy for the V operand
      int e = t + 256 * i;
      int key = e >> 4, dim = e & 15;
      Ktt[dim][key] = Kt[key][dim];
    }
    __syncthreads();

    v16h bK0 = half ? zero16h() : *(const v16h*)&Kt[l16][0];
    v16h bK1 = half ? zero16h() : *(const v16h*)&Kt[16 + l16][0];
    v8f z;
    for (int r = 0; r < 8; ++r) z[r] = 0.0f;
    v8f s0 = __builtin_amdgcn_wmma_f32_16x16x32_f16(false, aQ, false, bK0,
                                                    (short)0, z, false, false);
    v8f s1 = __builtin_amdgcn_wmma_f32_16x16x32_f16(false, aQ, false, bK1,
                                                    (short)0, z, false, false);
    for (int r = 0; r < 8; ++r) {
      float v = fmaxf(s0[r], s1[r]);
      for (int msk = 1; msk <= 8; msk <<= 1) v = fmaxf(v, __shfl_xor(v, msk, 32));
      float nm = fmaxf(rmax[r], v);
      float alpha = __expf(rmax[r] - nm);
      float p0 = __expf(s0[r] - nm), p1 = __expf(s1[r] - nm);
      float ls = p0 + p1;
      for (int msk = 1; msk <= 8; msk <<= 1) ls += __shfl_xor(ls, msk, 32);
      rsum[r] = rsum[r] * alpha + ls;
      rmax[r] = nm;
      oacc[r] *= alpha;
      int mrow = r + 8 * half;
      Pw[w][mrow][l16]      = (_Float16)p0;
      Pw[w][mrow][16 + l16] = (_Float16)p1;
    }
    __syncthreads();

    v8h lo = *(const v8h*)&Pw[w][l16][16 * half];
    v8h hi = *(const v8h*)&Pw[w][l16][16 * half + 8];
    v16h aP = cat8(lo, hi);
    v16h bV = *(const v16h*)&Ktt[l16][16 * half];
    oacc = __builtin_amdgcn_wmma_f32_16x16x32_f16(false, aP, false, bV,
                                                  (short)0, oacc, false, false);
  }

  for (int r = 0; r < 8; ++r) {
    int m = qt * 128 + w * 16 + r + 8 * half;
    O[(seqBase + m) * CDIM + h * 16 + l16] = (_Float16)(oacc[r] / rsum[r]);
  }
}

// ---------------------------------------------------------------------------
// Deterministic two-stage pooled mean + sigmoid gate (f32 branch outputs).
// ---------------------------------------------------------------------------
__global__ __launch_bounds__(128) void reduce_mean_kernel(
    const float* __restrict__ C1, const float* __restrict__ C2,
    float* __restrict__ partial) {
  const float* src = (blockIdx.y == 0) ? C1 : C2;
  int c = threadIdx.x;
  float sum = 0.0f;
  for (int i = 0; i < 256; ++i)
    sum += src[(size_t)(blockIdx.x * 256 + i) * CDIM + c];
  partial[((size_t)blockIdx.y * 124 + blockIdx.x) * CDIM + c] = sum;
}

__global__ __launch_bounds__(256) void gate_kernel(
    const float* __restrict__ partial, const float* __restrict__ Wg,
    const float* __restrict__ bg, float* __restrict__ gate) {
  __shared__ float g[256];
  int t = threadIdx.x;
  float sum = 0.0f;
  for (int i = 0; i < 124; ++i)
    sum += partial[((size_t)(t >> 7) * 124 + i) * CDIM + (t & 127)];
  g[t] = sum * (1.0f / (float)PPOS);
  __syncthreads();
  if (t < 128) {
    float acc = bg[t];
    for (int j = 0; j < 256; ++j) acc += g[j] * Wg[j * CDIM + t];
    gate[t] = 1.0f / (1.0f + __expf(-acc));
  }
}

// ---------------------------------------------------------------------------
// Gated fuse with LDS transpose: [p][c] branch outputs -> [c][p] output.
// ---------------------------------------------------------------------------
__global__ __launch_bounds__(256) void fuse_kernel(
    const float* __restrict__ C1, const float* __restrict__ C2,
    const float* __restrict__ gate, float* __restrict__ Y) {
  __shared__ float Ts[32][65], Tt[32][65];
  __shared__ float gs[32];
  const int p0 = blockIdx.x * 64, c0 = blockIdx.y * 32;
  const int t = threadIdx.x;
  if (t < 32) gs[t] = gate[c0 + t];
  for (int i = 0; i < 8; ++i) {
    int e = t + 256 * i;
    int cl = e & 31, pl = e >> 5;
    size_t src = (size_t)(p0 + pl) * CDIM + c0 + cl;
    Ts[cl][pl] = C1[src];
    Tt[cl][pl] = C2[src];
  }
  __syncthreads();
  for (int i = 0; i < 8; ++i) {
    int e = t + 256 * i;
    int pl = e & 63, cl = e >> 6;
    float gv = gs[cl];
    Y[(size_t)(c0 + cl) * PPOS + p0 + pl] =
        gv * Ts[cl][pl] + (1.0f - gv) * Tt[cl][pl];
  }
}

// ---------------------------------------------------------------------------
extern "C" void kernel_launch(void* const* d_in, const int* in_sizes, int n_in,
                              void* d_out, int out_size, void* d_ws,
                              size_t ws_size, hipStream_t stream) {
  (void)in_sizes; (void)n_in; (void)out_size; (void)ws_size;
  const float* x       = (const float*)d_in[0];
  const float* Wq_spec = (const float*)d_in[1];
  const float* bq_spec = (const float*)d_in[2];
  const float* Wp_spec = (const float*)d_in[3];
  const float* bp_spec = (const float*)d_in[4];
  const float* Wq_spat = (const float*)d_in[5];
  const float* bq_spat = (const float*)d_in[6];
  const float* Wp_spat = (const float*)d_in[7];
  const float* bp_spat = (const float*)d_in[8];
  const float* Wg      = (const float*)d_in[9];
  const float* bg      = (const float*)d_in[10];
  float* out = (float*)d_out;

  // Workspace (~57 MB): every word written before read.
  const size_t NPC = (size_t)PPOS * CDIM;
  char* wsp = (char*)d_ws;
  _Float16* xh   = (_Float16*)wsp;  wsp += NPC * 2;  // x as f16 [p][c]
  _Float16* bufQ = (_Float16*)wsp;  wsp += NPC * 2;  // Q projection
  _Float16* bufB = (_Float16*)wsp;  wsp += NPC * 2;  // attention output
  float* bufC1   = (float*)wsp;     wsp += NPC * 4;  // spectral branch out
  float* bufC2   = (float*)wsp;     wsp += NPC * 4;  // spatial branch out
  float* partial = (float*)wsp;     wsp += 2 * 124 * CDIM * 4;
  float* gate    = (float*)wsp;

  x_to_f16_kernel<<<dim3(496, 4), 256, 0, stream>>>(x, xh);
  // ---- spectral branch ----
  gemm128_kernel<true><<<248, 256, 0, stream>>>(xh, Wq_spec, bq_spec, bufQ);
  spectral_attn_kernel<<<1024, 256, 0, stream>>>(bufQ, bufB);
  gemm128_kernel<false><<<248, 256, 0, stream>>>(bufB, Wp_spec, bp_spec, bufC1);
  // ---- spatial branch ----
  gemm128_kernel<true><<<248, 256, 0, stream>>>(xh, Wq_spat, bq_spat, bufQ);
  spatial_flash_kernel<<<dim3(8, 8, 31), 256, 0, stream>>>(bufQ, bufB);
  gemm128_kernel<false><<<248, 256, 0, stream>>>(bufB, Wp_spat, bp_spat, bufC2);
  // ---- gate + fuse ----
  reduce_mean_kernel<<<dim3(124, 2), 128, 0, stream>>>(bufC1, bufC2, partial);
  gate_kernel<<<1, 256, 0, stream>>>(partial, Wg, bg, gate);
  fuse_kernel<<<dim3(496, 4), 256, 0, stream>>>(bufC1, bufC2, gate, out);
}
